// PatchInpainting_32796370272592
// MI455X (gfx1250) — matrix operand order repaired and checked by reading, not compiled, not run on hardware
//
#include <hip/hip_runtime.h>
#include <hip/hip_bf16.h>
#include <float.h>

typedef __attribute__((ext_vector_type(16))) _Float16 v16h;
typedef __attribute__((ext_vector_type(8)))  float    v8f;
typedef __attribute__((ext_vector_type(4)))  int      v4i;

#define AS1 __attribute__((address_space(1)))
#define AS3 __attribute__((address_space(3)))

// Async memory->LDS staging (gfx1250 GLOBAL_LOAD_ASYNC_TO_LDS_B128), guarded so
// a missing builtin falls back to plain loads instead of breaking the build.
#if defined(__HIP_DEVICE_COMPILE__) && defined(__gfx1250__) && \
    __has_builtin(__builtin_amdgcn_global_load_async_to_lds_b128)
#define ASYNC_LDS 1
#else
#define ASYNC_LDS 0
#endif

// Problem dims (fixed by the reference)
#define B_    8
#define L_    1024
#define E_    896
#define H_    4
#define DK_   2048
#define DV_   768
#define TOPK_ 64

// GEMM tiling: block = 256 threads = 8 wave32s in a 2(M) x 4(N) grid,
// each wave computes a 32x64 tile via 2x4 v_wmma_f32_16x16x32_f16
// (8 WMMAs per K-step, each A fragment reused 4x -> 2x LDS intensity vs 32x32).
constexpr int BM  = 64;
constexpr int BN  = 256;
constexpr int BK  = 32;
constexpr int PAD = 8;   // f16 elements of LDS row padding

// ---------------------------------------------------------------------------
// Templated batched GEMM:  C[z] = alpha * A[z] * op(B[z])
//   A is [M,K] row-major (f32 or f16).
//   B_NT=true : B is [N,K] row-major (C += A*B^T)   (projection / QK^T / FC)
//   B_NT=false: B is [K,N] row-major (C += A*B)     (attn * V)
//   z in gridDim.z decomposes as (b = z/Hn, h = z%Hn) with independent strides.
// ---------------------------------------------------------------------------
template<bool A_F16, bool B_F16, bool B_NT, bool C_F16>
__global__ __launch_bounds__(256)
void gemm_wmma_kernel(const void* __restrict__ Ap, const void* __restrict__ Bp,
                      void* __restrict__ Cp,
                      int M, int N, int K, int lda, int ldb, int ldc,
                      long aSB, long aSH, long bSB, long bSH, long cSB, long cSH,
                      int Hn, float alpha)
{
    __shared__ _Float16 As[BM][BK + PAD];
    __shared__ _Float16 Bs[BN][BK + PAD];

    const int zb = blockIdx.z / Hn;
    const int zh = blockIdx.z % Hn;
    const size_t aOff = (size_t)zb * aSB + (size_t)zh * aSH;
    const size_t bOff = (size_t)zb * bSB + (size_t)zh * bSH;
    const size_t cOff = (size_t)zb * cSB + (size_t)zh * cSH;

    const int tid  = threadIdx.x;
    const int lane = tid & 31;
    const int wave = tid >> 5;
    const int wm   = (wave & 1) * 32;   // wave M offset within block tile
    const int wn   = (wave >> 1) * 64;  // wave N offset within block tile
    const int m0   = blockIdx.y * BM;
    const int n0   = blockIdx.x * BN;

    v8f acc[2][4] = {};

    const int mrow  = lane & 15;          // M (or N) index within 16
    const int khalf = (lane >> 4) * 8;    // unused (fragments use K offset 0 or 16)

    for (int kt = 0; kt < K; kt += BK) {
        __syncthreads();

        // ---- prefetch next K-tile into L2 (global_prefetch_b8) ----
        if (kt + BK < K) {
            const size_t aeSz = A_F16 ? 2 : 4;
            const size_t beSz = B_F16 ? 2 : 4;
            __builtin_prefetch((const char*)Ap +
                (aOff + (size_t)(m0 + (tid & (BM - 1))) * lda + (kt + BK)) * aeSz, 0, 3);
            const size_t bIdx = B_NT
                ? bOff + (size_t)(n0 + tid) * ldb + (kt + BK)
                : bOff + (size_t)(kt + BK + (tid & (BK - 1))) * ldb + n0;
            __builtin_prefetch((const char*)Bp + bIdx * beSz, 0, 3);
        }

        // ---- stage A tile [BM x BK] as f16 ----
        if constexpr (A_F16) {
#if ASYNC_LDS
            {   // BM*BK/8 == 256 chunks of 16B -> exactly one per thread
                const int r  = tid >> 2;
                const int cc = (tid & 3) * 8;
                const _Float16* src = (const _Float16*)Ap + aOff +
                                      (size_t)(m0 + r) * lda + (kt + cc);
                __builtin_amdgcn_global_load_async_to_lds_b128(
                    (AS1 v4i*)src, (AS3 v4i*)&As[r][cc], 0, 0);
            }
#else
            for (int i = tid; i < BM * BK; i += 256) {
                const int r = i >> 5, c = i & 31;
                As[r][c] = ((const _Float16*)Ap)[aOff + (size_t)(m0 + r) * lda + (kt + c)];
            }
#endif
        } else {
            for (int i = tid; i < BM * BK; i += 256) {
                const int r = i >> 5, c = i & 31;
                As[r][c] = (_Float16)(((const float*)Ap)[aOff + (size_t)(m0 + r) * lda + (kt + c)]);
            }
        }

        // ---- stage B tile [BN x BK] as f16 (always [n][k] in LDS) ----
        if constexpr (B_F16 && B_NT) {
#if ASYNC_LDS
            for (int i = tid; i < (BN * BK) / 8; i += 256) {  // 4 chunks/thread
                const int n  = i >> 2;
                const int cc = (i & 3) * 8;
                const _Float16* src = (const _Float16*)Bp + bOff +
                                      (size_t)(n0 + n) * ldb + (kt + cc);
                __builtin_amdgcn_global_load_async_to_lds_b128(
                    (AS1 v4i*)src, (AS3 v4i*)&Bs[n][cc], 0, 0);
            }
#else
            for (int i = tid; i < BN * BK; i += 256) {
                const int n = i >> 5, c = i & 31;
                Bs[n][c] = ((const _Float16*)Bp)[bOff + (size_t)(n0 + n) * ldb + (kt + c)];
            }
#endif
        } else {
            for (int i = tid; i < BN * BK; i += 256) {
                int n, c;
                if constexpr (B_NT) { n = i >> 5; c = i & 31; }       // coalesced over K
                else                { c = i >> 8; n = i & (BN - 1); } // coalesced over N
                const size_t g = B_NT ? bOff + (size_t)(n0 + n) * ldb + (kt + c)
                                      : bOff + (size_t)(kt + c) * ldb + (n0 + n);
                if constexpr (B_F16) Bs[n][c] = ((const _Float16*)Bp)[g];
                else                 Bs[n][c] = (_Float16)(((const float*)Bp)[g]);
            }
        }

#if ASYNC_LDS
        if constexpr (A_F16 || (B_F16 && B_NT)) {
#if __has_builtin(__builtin_amdgcn_s_wait_asynccnt)
            __builtin_amdgcn_s_wait_asynccnt(0);
#else
            asm volatile("s_wait_asynccnt 0x0" ::: "memory");
#endif
        }
#endif
        __syncthreads();

        // ---- gather WMMA fragments from LDS ----
        // A 16x32 f16: lanes 0-15 hold M=0..15; halves j -> K = (lane>=16)*16 + j.
        v16h afrag[2], bfrag[4];
        #pragma unroll
        for (int t = 0; t < 2; ++t) {
            const _Float16* ap = &As[wm + t * 16 + mrow][(lane >> 4) * 16];
            #pragma unroll
            for (int j = 0; j < 16; ++j) afrag[t][j] = ap[j];
        }
        // B 32x16 f16: lane n holds column n; halves j -> K = (lane>=16)*16 + j.
        #pragma unroll
        for (int u = 0; u < 4; ++u) {
            const _Float16* bp = &Bs[wn + u * 16 + mrow][(lane >> 4) * 16];
            #pragma unroll
            for (int j = 0; j < 16; ++j) bfrag[u][j] = bp[j];
        }

        #pragma unroll
        for (int t = 0; t < 2; ++t)
            #pragma unroll
            for (int u = 0; u < 4; ++u)
                acc[t][u] = __builtin_amdgcn_wmma_f32_16x16x32_f16(
                    false, afrag[t], false, bfrag[u],
                    (short)0, acc[t][u], false, false);
    }

    // ---- epilogue: C/D layout: VGPR r -> M = r + (lane>=16)*8, N = lane&15 ----
    const int rofs = (lane >> 4) << 3;
    #pragma unroll
    for (int t = 0; t < 2; ++t)
        #pragma unroll
        for (int u = 0; u < 4; ++u)
            #pragma unroll
            for (int r = 0; r < 8; ++r) {
                const int row = m0 + wm + t * 16 + r + rofs;
                const int col = n0 + wn + u * 16 + mrow;
                const float val = acc[t][u][r] * alpha;
                const size_t g = cOff + (size_t)row * ldc + col;
                if constexpr (C_F16) ((_Float16*)Cp)[g] = (_Float16)val;
                else                 ((float*)Cp)[g] = val;
            }
}

// ---------------------------------------------------------------------------
// Top-64 keep-mask + softmax, in place on attn [B,H,L,L] (f32).
// One wave32 per row of 1024 scores; 32 values per lane in registers.
// Queries with mask <= 0.5 attend everywhere.
// ---------------------------------------------------------------------------
__global__ __launch_bounds__(256)
void topk_softmax_kernel(float* __restrict__ attn, const float* __restrict__ qmask)
{
    const int gwave = (int)((blockIdx.x * 256 + threadIdx.x) >> 5);
    const int lane  = threadIdx.x & 31;
    const int total = B_ * H_ * L_;
    if (gwave >= total) return;

    const int b = gwave / (H_ * L_);
    const int q = gwave % L_;
    float* row = attn + (size_t)gwave * L_;

    float vals[32];
    #pragma unroll
    for (int i = 0; i < 32; ++i) vals[i] = row[i * 32 + lane];

    const bool masked = qmask[b * L_ + q] > 0.5f;

    unsigned keep;
    if (masked) {
        keep = 0u;
        unsigned removed = 0u;
        for (int it = 0; it < TOPK_; ++it) {
            // local max among not-yet-removed values (bitmask, no dyn indexing)
            float mv = -FLT_MAX; int mi = 0;
            #pragma unroll
            for (int i = 0; i < 32; ++i) {
                const bool ok = ((removed >> i) & 1u) == 0u;
                if (ok && vals[i] > mv) { mv = vals[i]; mi = i; }
            }
            // wave32 argmax (ties -> lowest lane)
            float bv = mv; int bl = lane;
            #pragma unroll
            for (int off = 16; off > 0; off >>= 1) {
                const float ov = __shfl_xor(bv, off, 32);
                const int   ol = __shfl_xor(bl, off, 32);
                if (ov > bv || (ov == bv && ol < bl)) { bv = ov; bl = ol; }
            }
            if (bl == lane) { keep |= 1u << mi; removed |= 1u << mi; }
        }
    } else {
        keep = 0xFFFFFFFFu;
    }

    // masked softmax
    float rmax = -FLT_MAX;
    #pragma unroll
    for (int i = 0; i < 32; ++i) {
        const float v = ((keep >> i) & 1u) ? vals[i] : -FLT_MAX;
        vals[i] = v;
        rmax = fmaxf(rmax, v);
    }
    for (int off = 16; off > 0; off >>= 1) rmax = fmaxf(rmax, __shfl_xor(rmax, off, 32));
    float s = 0.0f;
    #pragma unroll
    for (int i = 0; i < 32; ++i) {
        const float e = __expf(vals[i] - rmax);
        vals[i] = e;
        s += e;
    }
    for (int off = 16; off > 0; off >>= 1) s += __shfl_xor(s, off, 32);
    const float inv = 1.0f / s;
    #pragma unroll
    for (int i = 0; i < 32; ++i) row[i * 32 + lane] = vals[i] * inv;
}

// ---------------------------------------------------------------------------
extern "C" void kernel_launch(void* const* d_in, const int* in_sizes, int n_in,
                              void* d_out, int out_size, void* d_ws, size_t ws_size,
                              hipStream_t stream)
{
    (void)in_sizes; (void)n_in; (void)out_size; (void)ws_size;

    const float* q   = (const float*)d_in[0];
    const float* k   = (const float*)d_in[1];
    const float* v   = (const float*)d_in[2];
    const float* qm  = (const float*)d_in[3];
    const float* wqs = (const float*)d_in[4];
    const float* wks = (const float*)d_in[5];
    const float* wvs = (const float*)d_in[6];
    const float* fcw = (const float*)d_in[7];

    float* out  = (float*)d_out;                              // [B, L, DV]
    float* attn = out + (size_t)B_ * L_ * DV_;                // [B, H, L, L]

    // Workspace: f16 q_proj / k_proj / v_proj / mixed
    _Float16* qp = (_Float16*)d_ws;                           // [B*L, H*DK]
    _Float16* kp = qp + (size_t)B_ * L_ * H_ * DK_;           // [B*L, H*DK]
    _Float16* vp = kp + (size_t)B_ * L_ * H_ * DK_;           // [B*L, H*DV]
    _Float16* mx = vp + (size_t)B_ * L_ * H_ * DV_;           // [B*L, H*DV]

    const dim3 blk(256);
    const int ML = B_ * L_;             // 8192 rows
    const float iscale = 0.022097086912079608f;  // 1/sqrt(DK)

    // 1) q_proj = q @ w_qs^T        [8192x896]x[8192x896]^T -> f16 [8192x8192]
    gemm_wmma_kernel<false, false, true, true>
        <<<dim3((H_ * DK_) / BN, ML / BM, 1), blk, 0, stream>>>(
        q, wqs, qp, ML, H_ * DK_, E_, E_, E_, H_ * DK_,
        0, 0, 0, 0, 0, 0, 1, 1.0f);

    // 2) k_proj = k @ w_ks^T
    gemm_wmma_kernel<false, false, true, true>
        <<<dim3((H_ * DK_) / BN, ML / BM, 1), blk, 0, stream>>>(
        k, wks, kp, ML, H_ * DK_, E_, E_, E_, H_ * DK_,
        0, 0, 0, 0, 0, 0, 1, 1.0f);

    // 3) v_proj = v @ w_vs^T        [8192x768]x[3072x768]^T -> f16 [8192x3072]
    gemm_wmma_kernel<false, false, true, true>
        <<<dim3((H_ * DV_) / BN, ML / BM, 1), blk, 0, stream>>>(
        v, wvs, vp, ML, H_ * DV_, DV_, DV_, DV_, H_ * DV_,
        0, 0, 0, 0, 0, 0, 1, 1.0f);

    // 4) scores[b,h] = (1/sqrt(DK)) * Qh @ Kh^T   -> f32 attn region of d_out
    gemm_wmma_kernel<true, true, true, false>
        <<<dim3(L_ / BN, L_ / BM, B_ * H_), blk, 0, stream>>>(
        qp, kp, attn, L_, L_, DK_, H_ * DK_, H_ * DK_, L_,
        (long)L_ * H_ * DK_, (long)DK_,
        (long)L_ * H_ * DK_, (long)DK_,
        (long)H_ * L_ * L_,  (long)L_ * L_,
        H_, iscale);

    // 5) top-64 mask + softmax in place (one wave per row, 32768 rows)
    topk_softmax_kernel<<<dim3((B_ * H_ * L_) / 8), blk, 0, stream>>>(attn, qm);

    // 6) mixed[b,h] = attn @ Vh     (NN)  -> f16 mixed [B*L, H*DV]
    gemm_wmma_kernel<false, true, false, true>
        <<<dim3(DV_ / BN, L_ / BM, B_ * H_), blk, 0, stream>>>(
        attn, vp, mx, L_, DV_, L_, L_, H_ * DV_, H_ * DV_,
        (long)H_ * L_ * L_,  (long)L_ * L_,
        (long)L_ * H_ * DV_, (long)DV_,
        (long)L_ * H_ * DV_, (long)DV_,
        H_, 1.0f);

    // 7) out = mixed @ fc_w^T       [8192x3072]x[768x3072]^T -> f32 [8192x768]
    gemm_wmma_kernel<true, false, true, false>
        <<<dim3(DV_ / BN, ML / BM, 1), blk, 0, stream>>>(
        mx, fcw, out, ML, DV_, H_ * DV_, H_ * DV_, H_ * DV_, DV_,
        0, 0, 0, 0, 0, 0, 1, 1.0f);
}
